// RN_44598940402016
// MI455X (gfx1250) — matrix operand order repaired and checked by reading, not compiled, not run on hardware
//
#include <hip/hip_runtime.h>
#include <stdint.h>

typedef __attribute__((ext_vector_type(16))) __bf16 v16bf;
typedef __attribute__((ext_vector_type(8)))  float  v8f;

__device__ __forceinline__ unsigned short f2bf(float f) {
  union { float f; unsigned u; } x; x.f = f;
  unsigned r = x.u + 0x7FFFu + ((x.u >> 16) & 1u);
  return (unsigned short)(r >> 16);
}
__device__ __forceinline__ float bf2f(unsigned short h) {
  union { unsigned u; float f; } x; x.u = ((unsigned)h) << 16;
  return x.f;
}

// ---------------- Conv 3x3 stride2 pad1 + bias + ReLU ----------------
__global__ void conv3x3_s2_relu(const float* __restrict__ in,
                                const float* __restrict__ w,
                                const float* __restrict__ bias,
                                float* __restrict__ out,
                                int B_, int Cin, int Hin, int Win, int Cout) {
  int Ho = Hin / 2, Wo = Win / 2;
  size_t idx = (size_t)blockIdx.x * blockDim.x + threadIdx.x;
  size_t total = (size_t)B_ * Cout * Ho * Wo;
  if (idx >= total) return;
  int wo = idx % Wo; size_t t = idx / Wo;
  int ho = t % Ho; t /= Ho;
  int co = t % Cout; int bb = (int)(t / Cout);
  float s = bias[co];
  const float* wp = w + (size_t)co * Cin * 9;
  const float* ip = in + (size_t)bb * Cin * Hin * Win;
  for (int ci = 0; ci < Cin; ci++) {
    const float* ic = ip + (size_t)ci * Hin * Win;
    const float* wc = wp + ci * 9;
    #pragma unroll
    for (int kh = 0; kh < 3; kh++) {
      int hi = 2 * ho - 1 + kh;
      if ((unsigned)hi >= (unsigned)Hin) continue;
      #pragma unroll
      for (int kw = 0; kw < 3; kw++) {
        int wi = 2 * wo - 1 + kw;
        if ((unsigned)wi >= (unsigned)Win) continue;
        s += ic[hi * Win + wi] * wc[kh * 3 + kw];
      }
    }
  }
  out[idx] = fmaxf(s, 0.f);
}

// ---------------- BatchNorm (training batch stats) ----------------
__global__ void bn_stats(const float* __restrict__ x, float* __restrict__ mean,
                         float* __restrict__ rstd, int C, int HW, int Btot) {
  int c = blockIdx.x;
  __shared__ float ssum[256], ssq[256];
  float s = 0.f, q = 0.f;
  int n = Btot * HW;
  for (int i = threadIdx.x; i < n; i += blockDim.x) {
    int bb = i / HW, p = i % HW;
    float v = x[((size_t)bb * C + c) * HW + p];
    s += v; q += v * v;
  }
  ssum[threadIdx.x] = s; ssq[threadIdx.x] = q;
  __syncthreads();
  for (int off = 128; off > 0; off >>= 1) {
    if (threadIdx.x < off) {
      ssum[threadIdx.x] += ssum[threadIdx.x + off];
      ssq[threadIdx.x]  += ssq[threadIdx.x + off];
    }
    __syncthreads();
  }
  if (threadIdx.x == 0) {
    float fn = (float)n;
    float m = ssum[0] / fn;
    float v = ssq[0] / fn - m * m;
    mean[c] = m;
    rstd[c] = rsqrtf(v + 1e-5f);
  }
}

__global__ void bn_apply(float* __restrict__ x, const float* __restrict__ mean,
                         const float* __restrict__ rstd, const float* __restrict__ g,
                         const float* __restrict__ beta, int C, int HW, size_t total) {
  size_t i = (size_t)blockIdx.x * blockDim.x + threadIdx.x;
  if (i >= total) return;
  int c = (int)((i / HW) % C);
  x[i] = (x[i] - mean[c]) * rstd[c] * g[c] + beta[c];
}

// ---------------- feature build: [128,256,5,5] -> bf16 [3200, 288] ----------------
__global__ void build_features(const float* __restrict__ x4, unsigned short* __restrict__ Xf) {
  int idx = blockIdx.x * blockDim.x + threadIdx.x;
  if (idx >= 3200 * 288) return;
  int col = idx % 288; int row = idx / 288;
  int b = row / 25, p = row % 25;
  float v;
  if (col < 256)      v = x4[((size_t)b * 256 + col) * 25 + p];
  else if (col == 256) v = (float)(p / 5) * 0.5f - 1.f;
  else if (col == 257) v = (float)(p % 5) * 0.5f - 1.f;
  else                 v = 0.f;
  Xf[idx] = f2bf(v);
}

// ---------------- fp32 [rows, stride] (+colOff) -> zero-padded bf16 [dstRows, Kpad] ----------------
__global__ void convert_w(const float* __restrict__ src, int srcRows, int srcStride,
                          int colOff, int Kreal,
                          unsigned short* __restrict__ dst, int dstRows, int Kpad) {
  size_t idx = (size_t)blockIdx.x * blockDim.x + threadIdx.x;
  size_t total = (size_t)dstRows * Kpad;
  if (idx >= total) return;
  int k = (int)(idx % Kpad); int n = (int)(idx / Kpad);
  float v = (n < srcRows && k < Kreal) ? src[(size_t)n * srcStride + colOff + k] : 0.f;
  dst[idx] = f2bf(v);
}

// ---------------- Qv[b,n] = g1b[n] + q[b] . g1w[n, 516:527] ----------------
__global__ void qv_kernel(const float* __restrict__ q, const float* __restrict__ g1w,
                          const float* __restrict__ g1b, float* __restrict__ Qv) {
  int idx = blockIdx.x * blockDim.x + threadIdx.x;
  if (idx >= 128 * 2048) return;
  int n = idx % 2048, b = idx / 2048;
  float s = 0.f;
  if (n < 2000) {
    s = g1b[n];
    #pragma unroll
    for (int k = 0; k < 11; k++) s += q[b * 11 + k] * g1w[(size_t)n * 527 + 516 + k];
  }
  Qv[idx] = s;
}

// ---------------- WMMA GEMM: out[M,Npad] = A[M,K]bf16 @ W[Npad,K]bf16^T ----------------
// Async Global->LDS staging (ASYNCcnt), double-buffered LDS, bf16 WMMA w/ fp32 accum.
union FragBF { v16bf v; uint4 q[2]; };

__global__ void __launch_bounds__(256)
wmma_gemm(const unsigned short* __restrict__ A, int lda,
          const unsigned short* __restrict__ W, int ldw,
          const float* __restrict__ bias, int Nreal,
          void* __restrict__ outp, int ldo, int Kpad,
          int relu, int out_bf16) {
  // double-buffered tiles: 2 x (128 rows x 32 bf16) for A and for B = 32 KB
  __shared__ __align__(16) unsigned short As[2 * 128 * 32];
  __shared__ __align__(16) unsigned short Bs[2 * 128 * 32];

  const int tid  = threadIdx.x;
  const int wave = tid >> 5;
  const int lane = tid & 31;
  const int half = lane >> 4;
  const int l15  = lane & 15;
  const int wm = (wave & 1) * 64;       // wave M base in block (2 waves along M)
  const int wn = (wave >> 1) * 32;      // wave N base in block (4 waves along N)
  const int m0 = blockIdx.y * 128;
  const int n0 = blockIdx.x * 128;

  v8f zero = {0.f, 0.f, 0.f, 0.f, 0.f, 0.f, 0.f, 0.f};
  v8f acc[4][2];
  #pragma unroll
  for (int i = 0; i < 4; i++)
    #pragma unroll
    for (int j = 0; j < 2; j++) acc[i][j] = zero;

  // cooperative staging mapping: 2 threads per tile-row, one 16B (8 bf16) load each, at col 0 or 16
  const int cr = tid >> 1;
  const int ch = (tid & 1) * 16;
  const unsigned short* aRow = A + (size_t)(m0 + cr) * lda + ch;
  const unsigned short* wRow = W + (size_t)(n0 + cr) * ldw + ch;
  const unsigned asOff0 = (unsigned)(size_t)(As + cr * 32 + ch);  // LDS byte offset
  const unsigned bsOff0 = (unsigned)(size_t)(Bs + cr * 32 + ch);

  const int nt = Kpad >> 5;

  // issue one tile's async DMA (A + B halves owned by this thread)
  auto issue = [&](int tile, int buf) {
    const unsigned short* ap = aRow + tile * 32;
    const unsigned short* bp = wRow + tile * 32;
    unsigned ao = asOff0 + buf * 8192;
    unsigned bo = bsOff0 + buf * 8192;
    asm volatile(
        "global_load_async_to_lds_b128 %0, %2, off\n\t"
        "global_load_async_to_lds_b128 %1, %3, off"
        :: "v"(ao), "v"(bo), "v"(ap), "v"(bp)
        : "memory");
  };

  issue(0, 0);
  for (int kt = 0; kt < nt; ++kt) {
    // prefetch next tile into the other buffer (dummy re-issue on the last
    // iteration keeps ASYNCcnt bookkeeping uniform; it targets the dead buffer)
    int nxt = (kt + 1 < nt) ? (kt + 1) : kt;
    issue(nxt, (kt + 1) & 1);
    // wait for current tile only (ASYNCcnt completes in order): <=2 leaves the
    // two just-issued prefetch ops in flight under the WMMA phase
    asm volatile("s_wait_asynccnt 0x2" ::: "memory");
    __syncthreads();

    const unsigned short* AsB = As + (kt & 1) * 4096;
    const unsigned short* BsB = Bs + (kt & 1) * 4096;

    FragBF af[4], bfm[2];
    #pragma unroll
    for (int mt = 0; mt < 4; mt++) {
      const unsigned short* p = AsB + (wm + mt * 16 + l15) * 32;
      af[mt].q[0] = *(const uint4*)(p + half * 8);        // K = h*8 .. h*8+7
      af[mt].q[1] = *(const uint4*)(p + 16 + half * 8);   // K = 16+h*8 .. +7
    }
    #pragma unroll
    for (int nt2 = 0; nt2 < 2; nt2++) {
      const unsigned short* p = BsB + (wn + nt2 * 16 + l15) * 32; // col=l15, K=h*8..+7, 16+h*8..+7
      bfm[nt2].q[0] = *(const uint4*)(p + half * 8);
      bfm[nt2].q[1] = *(const uint4*)(p + 16 + half * 8);
    }
    #pragma unroll
    for (int mt = 0; mt < 4; mt++)
      #pragma unroll
      for (int nt2 = 0; nt2 < 2; nt2++)
        acc[mt][nt2] = __builtin_amdgcn_wmma_f32_16x16x32_bf16(
            false, af[mt].v, false, bfm[nt2].v, (short)0, acc[mt][nt2], false, false);

    __syncthreads();  // all waves done reading buf (kt&1) before it is re-filled
  }
  asm volatile("s_wait_asynccnt 0x0" ::: "memory");

  // epilogue: D layout -> lane holds col n=l15, rows m = half*8 + v
  #pragma unroll
  for (int mt = 0; mt < 4; mt++) {
    #pragma unroll
    for (int nt2 = 0; nt2 < 2; nt2++) {
      int n = n0 + wn + nt2 * 16 + l15;
      if (n >= ldo) continue;
      float bv = (bias != nullptr && n < Nreal) ? bias[n] : 0.f;
      #pragma unroll
      for (int v = 0; v < 8; v++) {
        int m = m0 + wm + mt * 16 + half * 8 + v;
        float x = acc[mt][nt2][v] + bv;
        if (relu) x = fmaxf(x, 0.f);
        if (n >= Nreal) x = 0.f;
        if (out_bf16) ((unsigned short*)outp)[(size_t)m * ldo + n] = f2bf(x);
        else          ((float*)outp)[(size_t)m * ldo + n] = x;
      }
    }
  }
}

// ---------------- h1[b,i,j,:] = relu(A[b,j] + C[b,i] + Qv[b]) ----------------
__global__ void combine_h1(const float* __restrict__ Abuf, const float* __restrict__ Cbuf,
                           const float* __restrict__ Qv, unsigned short* __restrict__ H) {
  size_t idx = ((size_t)blockIdx.x * blockDim.x + threadIdx.x) * 4;
  if (idx >= (size_t)80000 * 2048) return;
  size_t row = idx / 2048; int n = (int)(idx % 2048);
  int b = (int)(row / 625); int ij = (int)(row % 625);
  int i = ij / 25, j = ij % 25;
  const float* ap = Abuf + ((size_t)(b * 25 + j)) * 2048 + n;
  const float* cp = Cbuf + ((size_t)(b * 25 + i)) * 2048 + n;
  const float* qp = Qv + (size_t)b * 2048 + n;
  unsigned short* hp = H + idx;
  #pragma unroll
  for (int u = 0; u < 4; u++) hp[u] = f2bf(fmaxf(ap[u] + cp[u] + qp[u], 0.f));
}

// ---------------- S[b,n] = sum_{p<625} H[b,p,n] (bf16 out) ----------------
__global__ void sum_pairs(const unsigned short* __restrict__ H, unsigned short* __restrict__ Sb) {
  int idx = blockIdx.x * blockDim.x + threadIdx.x;
  if (idx >= 128 * 2048) return;
  int n = idx % 2048, b = idx / 2048;
  const unsigned short* p = H + (size_t)b * 625 * 2048 + n;
  float s = 0.f;
  for (int t = 0; t < 625; t++) s += bf2f(p[(size_t)t * 2048]);
  Sb[idx] = f2bf(s);
}

// ---------------- final: relu(F4 @ ow^T + ob) -> log_softmax ----------------
__global__ void final_head(const float* __restrict__ F4, const float* __restrict__ ow,
                           const float* __restrict__ ob, float* __restrict__ out) {
  int b = blockIdx.x * blockDim.x + threadIdx.x;
  if (b >= 128) return;
  float z[10]; float mx = -1e30f;
  for (int c = 0; c < 10; c++) {
    float s = ob[c];
    for (int k = 0; k < 100; k++) s += F4[(size_t)b * 128 + k] * ow[c * 100 + k];
    s = fmaxf(s, 0.f);
    z[c] = s; mx = fmaxf(mx, s);
  }
  float se = 0.f;
  for (int c = 0; c < 10; c++) se += expf(z[c] - mx);
  float lse = mx + logf(se);
  for (int c = 0; c < 10; c++) out[b * 10 + c] = z[c] - lse;
}

// =====================================================================
extern "C" void kernel_launch(void* const* d_in, const int* in_sizes, int n_in,
                              void* d_out, int out_size, void* d_ws, size_t ws_size,
                              hipStream_t stream) {
  (void)in_sizes; (void)n_in; (void)out_size; (void)ws_size;
  const float* image     = (const float*)d_in[0];
  const float* questions = (const float*)d_in[1];
  const float* c1w = (const float*)d_in[2];  const float* c1b = (const float*)d_in[3];
  const float* bn1g = (const float*)d_in[4]; const float* bn1b = (const float*)d_in[5];
  const float* c2w = (const float*)d_in[6];  const float* c2b = (const float*)d_in[7];
  const float* bn2g = (const float*)d_in[8]; const float* bn2b = (const float*)d_in[9];
  const float* c3w = (const float*)d_in[10]; const float* c3b = (const float*)d_in[11];
  const float* bn3g = (const float*)d_in[12];const float* bn3b = (const float*)d_in[13];
  const float* c4w = (const float*)d_in[14]; const float* c4b = (const float*)d_in[15];
  const float* bn4g = (const float*)d_in[16];const float* bn4b = (const float*)d_in[17];
  const float* g1w = (const float*)d_in[18]; const float* g1b = (const float*)d_in[19];
  const float* g2w = (const float*)d_in[20]; const float* g2b = (const float*)d_in[21];
  const float* g3w = (const float*)d_in[22]; const float* g3b = (const float*)d_in[23];
  const float* g4w = (const float*)d_in[24]; const float* g4b = (const float*)d_in[25];
  const float* f1w = (const float*)d_in[26]; const float* f1b = (const float*)d_in[27];
  const float* f2w = (const float*)d_in[28]; const float* f2b = (const float*)d_in[29];
  const float* f3w = (const float*)d_in[30]; const float* f3b = (const float*)d_in[31];
  const float* f4w = (const float*)d_in[32]; const float* f4b = (const float*)d_in[33];
  const float* ow  = (const float*)d_in[34]; const float* ob  = (const float*)d_in[35];

  char* ws = (char*)d_ws;
  size_t off = 0;
  auto alloc = [&](size_t bytes) -> void* {
    void* p = ws + off; off += (bytes + 255) & ~(size_t)255; return p;
  };

  float* x1 = (float*)alloc((size_t)128 * 32 * 40 * 40 * 4);
  float* x2 = (float*)alloc((size_t)128 * 64 * 20 * 20 * 4);
  float* x3 = (float*)alloc((size_t)128 * 128 * 10 * 10 * 4);
  float* x4 = (float*)alloc((size_t)128 * 256 * 5 * 5 * 4);
  float* mean = (float*)alloc(256 * 4);
  float* rstd = (float*)alloc(256 * 4);
  unsigned short* Xf   = (unsigned short*)alloc((size_t)3200 * 288 * 2);
  unsigned short* Wo1b = (unsigned short*)alloc((size_t)2048 * 288 * 2);
  unsigned short* Wo2b = (unsigned short*)alloc((size_t)2048 * 288 * 2);
  unsigned short* g2wb = (unsigned short*)alloc((size_t)2048 * 2048 * 2);
  unsigned short* g3wb = (unsigned short*)alloc((size_t)2048 * 2048 * 2);
  unsigned short* g4wb = (unsigned short*)alloc((size_t)2048 * 2048 * 2);
  unsigned short* f1wb = (unsigned short*)alloc((size_t)2048 * 2048 * 2);
  unsigned short* f2wb = (unsigned short*)alloc((size_t)1024 * 2048 * 2);
  unsigned short* f3wb = (unsigned short*)alloc((size_t)512 * 1024 * 2);
  unsigned short* f4wb = (unsigned short*)alloc((size_t)128 * 512 * 2);
  float* Qv   = (float*)alloc((size_t)128 * 2048 * 4);
  float* Abuf = (float*)alloc((size_t)3200 * 2048 * 4);
  float* Cbuf = (float*)alloc((size_t)3200 * 2048 * 4);
  unsigned short* Hp0 = (unsigned short*)alloc((size_t)80000 * 2048 * 2);
  unsigned short* Hp1 = (unsigned short*)alloc((size_t)80000 * 2048 * 2);
  unsigned short* Sb  = (unsigned short*)alloc((size_t)128 * 2048 * 2);
  unsigned short* F1  = (unsigned short*)alloc((size_t)128 * 2048 * 2);
  unsigned short* F2  = (unsigned short*)alloc((size_t)128 * 1024 * 2);
  unsigned short* F3  = (unsigned short*)alloc((size_t)128 * 512 * 2);
  float* F4 = (float*)alloc((size_t)128 * 128 * 4);

  auto cdiv = [](size_t a, size_t b) -> unsigned { return (unsigned)((a + b - 1) / b); };
  dim3 blk(256);

  // ---- conv stack (each: conv+relu, batch stats, normalize) ----
  size_t n1 = (size_t)128 * 32 * 40 * 40;
  conv3x3_s2_relu<<<cdiv(n1, 256), blk, 0, stream>>>(image, c1w, c1b, x1, 128, 3, 80, 80, 32);
  bn_stats<<<32, blk, 0, stream>>>(x1, mean, rstd, 32, 1600, 128);
  bn_apply<<<cdiv(n1, 256), blk, 0, stream>>>(x1, mean, rstd, bn1g, bn1b, 32, 1600, n1);

  size_t n2 = (size_t)128 * 64 * 20 * 20;
  conv3x3_s2_relu<<<cdiv(n2, 256), blk, 0, stream>>>(x1, c2w, c2b, x2, 128, 32, 40, 40, 64);
  bn_stats<<<64, blk, 0, stream>>>(x2, mean, rstd, 64, 400, 128);
  bn_apply<<<cdiv(n2, 256), blk, 0, stream>>>(x2, mean, rstd, bn2g, bn2b, 64, 400, n2);

  size_t n3 = (size_t)128 * 128 * 10 * 10;
  conv3x3_s2_relu<<<cdiv(n3, 256), blk, 0, stream>>>(x2, c3w, c3b, x3, 128, 64, 20, 20, 128);
  bn_stats<<<128, blk, 0, stream>>>(x3, mean, rstd, 128, 100, 128);
  bn_apply<<<cdiv(n3, 256), blk, 0, stream>>>(x3, mean, rstd, bn3g, bn3b, 128, 100, n3);

  size_t n4 = (size_t)128 * 256 * 5 * 5;
  conv3x3_s2_relu<<<cdiv(n4, 256), blk, 0, stream>>>(x3, c4w, c4b, x4, 128, 128, 10, 10, 256);
  bn_stats<<<256, blk, 0, stream>>>(x4, mean, rstd, 256, 25, 128);
  bn_apply<<<cdiv(n4, 256), blk, 0, stream>>>(x4, mean, rstd, bn4g, bn4b, 256, 25, n4);

  // ---- features + weight conversion ----
  build_features<<<cdiv((size_t)3200 * 288, 256), blk, 0, stream>>>(x4, Xf);

  convert_w<<<cdiv((size_t)2048 * 288, 256), blk, 0, stream>>>(g1w, 2000, 527, 0,   258, Wo1b, 2048, 288);
  convert_w<<<cdiv((size_t)2048 * 288, 256), blk, 0, stream>>>(g1w, 2000, 527, 258, 258, Wo2b, 2048, 288);
  convert_w<<<cdiv((size_t)2048 * 2048, 256), blk, 0, stream>>>(g2w, 2000, 2000, 0, 2000, g2wb, 2048, 2048);
  convert_w<<<cdiv((size_t)2048 * 2048, 256), blk, 0, stream>>>(g3w, 2000, 2000, 0, 2000, g3wb, 2048, 2048);
  convert_w<<<cdiv((size_t)2048 * 2048, 256), blk, 0, stream>>>(g4w, 2000, 2000, 0, 2000, g4wb, 2048, 2048);
  convert_w<<<cdiv((size_t)2048 * 2048, 256), blk, 0, stream>>>(f1w, 2000, 2000, 0, 2000, f1wb, 2048, 2048);
  convert_w<<<cdiv((size_t)1024 * 2048, 256), blk, 0, stream>>>(f2w, 1000, 2000, 0, 2000, f2wb, 1024, 2048);
  convert_w<<<cdiv((size_t)512 * 1024, 256), blk, 0, stream>>>(f3w, 500, 1000, 0, 1000, f3wb, 512, 1024);
  convert_w<<<cdiv((size_t)128 * 512, 256), blk, 0, stream>>>(f4w, 100, 500, 0, 500, f4wb, 128, 512);

  qv_kernel<<<cdiv((size_t)128 * 2048, 256), blk, 0, stream>>>(questions, g1w, g1b, Qv);

  // ---- g1 decomposed: A = Xf@Wo1^T, C = Xf@Wo2^T (fp32 out), then pairwise combine ----
  wmma_gemm<<<dim3(16, 25), blk, 0, stream>>>(Xf, 288, Wo1b, 288, nullptr, 2000, Abuf, 2048, 288, 0, 0);
  wmma_gemm<<<dim3(16, 25), blk, 0, stream>>>(Xf, 288, Wo2b, 288, nullptr, 2000, Cbuf, 2048, 288, 0, 0);
  combine_h1<<<cdiv((size_t)80000 * 2048 / 4, 256), blk, 0, stream>>>(Abuf, Cbuf, Qv, Hp0);

  // ---- g2..g4: the 1.92 TFLOP WMMA core ----
  wmma_gemm<<<dim3(16, 625), blk, 0, stream>>>(Hp0, 2048, g2wb, 2048, g2b, 2000, Hp1, 2048, 2048, 1, 1);
  wmma_gemm<<<dim3(16, 625), blk, 0, stream>>>(Hp1, 2048, g3wb, 2048, g3b, 2000, Hp0, 2048, 2048, 1, 1);
  wmma_gemm<<<dim3(16, 625), blk, 0, stream>>>(Hp0, 2048, g4wb, 2048, g4b, 2000, Hp1, 2048, 2048, 1, 1);

  // ---- sum over 625 pairs, then f-MLP ----
  sum_pairs<<<cdiv((size_t)128 * 2048, 256), blk, 0, stream>>>(Hp1, Sb);
  wmma_gemm<<<dim3(16, 1), blk, 0, stream>>>(Sb, 2048, f1wb, 2048, f1b, 2000, F1, 2048, 2048, 1, 1);
  wmma_gemm<<<dim3(8, 1),  blk, 0, stream>>>(F1, 2048, f2wb, 2048, f2b, 1000, F2, 1024, 2048, 1, 1);
  wmma_gemm<<<dim3(4, 1),  blk, 0, stream>>>(F2, 1024, f3wb, 1024, f3b, 500,  F3, 512, 1024, 1, 1);
  wmma_gemm<<<dim3(1, 1),  blk, 0, stream>>>(F3, 512,  f4wb, 512,  f4b, 100,  F4, 128, 512, 1, 0);

  final_head<<<1, 128, 0, stream>>>(F4, ow, ob, (float*)d_out);
}
